// MultiHeadLatentAttention_13005160972822
// MI455X (gfx1250) — compile-verified, matrix operand order checked by reading, not run on hardware
//
#include <hip/hip_runtime.h>
#include <hip/hip_bf16.h>

typedef __attribute__((ext_vector_type(16))) _Float16 v16h;
typedef __attribute__((ext_vector_type(8)))  _Float16 v8h;
typedef __attribute__((ext_vector_type(8)))  float    v8f;

#define DMODEL   1024
#define NHEADS   16
#define HDIM     64
#define DLATENT  256
#define BATCH    4
#define SEQ      2048
#define MTOT     (BATCH * SEQ)   // 8192 token rows

static __device__ __forceinline__ v16h combine8(v8h lo, v8h hi) {
    return __builtin_shufflevector(lo, hi, 0,1,2,3,4,5,6,7,8,9,10,11,12,13,14,15);
}

static __device__ __forceinline__ v8f wmma_f16(v16h a, v16h b, v8f c) {
    return __builtin_amdgcn_wmma_f32_16x16x32_f16(false, a, false, b,
                                                  (short)0, c, false, false);
}

// ---------------------------------------------------------------------------
// f32 -> f16 conversion
// ---------------------------------------------------------------------------
__global__ void f32_to_f16_kernel(const float* __restrict__ in,
                                  _Float16* __restrict__ out, int n) {
    int i = blockIdx.x * blockDim.x + threadIdx.x;
    if (i < n) out[i] = (_Float16)in[i];
}

// ---------------------------------------------------------------------------
// WMMA GEMM:  C[M,N] = A[M,K] (row-major f16)  x  B[N,K]^T (row-major f16)
// Each wave computes a 16x64 strip: ONE A-fragment feeds FOUR independent
// WMMA accumulator chains per K=32 step (10 b128 loads -> 4 WMMAs).
// OUTMODE: 0 = f16 row-major, 1 = f32 row-major,
//          2 = f16 transposed for V ([B,H,HDIM,SEQ], packed v8h stores)
// ---------------------------------------------------------------------------
template<int OUTMODE>
__global__ void gemm_wmma_kernel(const _Float16* __restrict__ A,
                                 const _Float16* __restrict__ B,
                                 void* __restrict__ Cout,
                                 int M, int N, int K) {
    const int lane = threadIdx.x & 31;
    const int wave = threadIdx.x >> 5;
    const int m0   = blockIdx.x * 16;
    const int n0   = (blockIdx.y * (blockDim.x >> 5) + wave) * 64;
    const int hlf  = lane >> 4;
    const int l16  = lane & 15;
    const int kb   = hlf * 8;

    const _Float16* arow = A + (size_t)(m0 + l16) * K + kb;
    const _Float16* br0  = B + (size_t)(n0      + l16) * K + kb;
    const _Float16* br1  = B + (size_t)(n0 + 16 + l16) * K + kb;
    const _Float16* br2  = B + (size_t)(n0 + 32 + l16) * K + kb;
    const _Float16* br3  = B + (size_t)(n0 + 48 + l16) * K + kb;

    v8f acc[4] = {{}, {}, {}, {}};
    for (int k0 = 0; k0 < K; k0 += 32) {
        v16h af  = combine8(*(const v8h*)(arow + k0), *(const v8h*)(arow + k0 + 16));
        v16h bf0 = combine8(*(const v8h*)(br0 + k0),  *(const v8h*)(br0 + k0 + 16));
        v16h bf1 = combine8(*(const v8h*)(br1 + k0),  *(const v8h*)(br1 + k0 + 16));
        v16h bf2 = combine8(*(const v8h*)(br2 + k0),  *(const v8h*)(br2 + k0 + 16));
        v16h bf3 = combine8(*(const v8h*)(br3 + k0),  *(const v8h*)(br3 + k0 + 16));
        acc[0] = wmma_f16(af, bf0, acc[0]);
        acc[1] = wmma_f16(af, bf1, acc[1]);
        acc[2] = wmma_f16(af, bf2, acc[2]);
        acc[3] = wmma_f16(af, bf3, acc[3]);
    }

    #pragma unroll
    for (int t = 0; t < 4; ++t) {
        if (OUTMODE == 2) {
            // V^T store: vt[((b*NHEADS+h)*HDIM+d)*SEQ + token]
            const int n    = n0 + 16 * t + l16;
            const int hh   = n >> 6;          // head
            const int d    = n & 63;          // dim within head
            const int bidx = m0 >> 11;        // batch (SEQ = 2048)
            const int tt   = (m0 & 2047) + 8 * hlf;
            v8h pk;
            #pragma unroll
            for (int r = 0; r < 8; ++r) pk[r] = (_Float16)acc[t][r];
            _Float16* vt = (_Float16*)Cout;
            *(v8h*)(vt + (size_t)((bidx * NHEADS + hh) * HDIM + d) * SEQ + tt) = pk;
        } else {
            #pragma unroll
            for (int r = 0; r < 8; ++r) {
                const size_t idx = (size_t)(m0 + hlf * 8 + r) * N + (n0 + 16 * t + l16);
                if (OUTMODE == 1) ((float*)Cout)[idx] = acc[t][r];
                else              ((_Float16*)Cout)[idx] = (_Float16)acc[t][r];
            }
        }
    }
}

// ---------------------------------------------------------------------------
// Flash-attention (causal, online softmax), transposed-score formulation.
// grid = (SEQ/16/4, NHEADS, BATCH), block = 128 (4 waves, one q-tile each).
//
// S^T = K-tile x Q^T puts queries on lanes, keys on VGPR indices:
//  - per-query softmax stats = per-lane reduce + ONE shfl_xor(16) per stat
//  - exp(S^T) values sit exactly in the A-fragment layout needed for P*V
//    (lane = query, a[j]/a[j+8] = keys 8h+j / 8h+16+j) -> no LDS staging
//  - V is consumed pre-transposed ([B,H,D,T]) -> two 16B loads per fragment
// ---------------------------------------------------------------------------
__global__ void mla_attention_kernel(const _Float16* __restrict__ q,
                                     const _Float16* __restrict__ k,
                                     const _Float16* __restrict__ vT,
                                     _Float16* __restrict__ o) {
    const int lane  = threadIdx.x & 31;
    const int wave  = threadIdx.x >> 5;
    const int h     = blockIdx.y;
    const int b     = blockIdx.z;
    const int q0    = (blockIdx.x * (blockDim.x >> 5) + wave) * 16;
    const int hlf   = lane >> 4;
    const int l16   = lane & 15;
    const int kboff = hlf * 8;

    const _Float16* qbase  = q  + (size_t)(b * SEQ) * DMODEL + h * HDIM;
    const _Float16* kbase  = k  + (size_t)(b * SEQ) * DMODEL + h * HDIM;
    const _Float16* vtbase = vT + (size_t)((b * NHEADS + h) * HDIM) * SEQ;

    // Q as B operand (dims = K-dim, queries = N): contiguous over head dim
    const _Float16* qrow = qbase + (size_t)(q0 + l16) * DMODEL;
    v16h qb0 = combine8(*(const v8h*)(qrow + kboff),
                        *(const v8h*)(qrow + kboff + 16));
    v16h qb1 = combine8(*(const v8h*)(qrow + 32 + kboff),
                        *(const v8h*)(qrow + 32 + kboff + 16));

    float mrun = -3.0e38f;     // running max for query q0 + l16
    float lrun = 0.0f;         // running sum for query q0 + l16
    v8f acc[4] = {{}, {}, {}, {}};
    const float scale = 0.125f;             // 1/sqrt(64)
    const int   qidx  = q0 + l16;
    const int   nkt   = (q0 + 47) / 32;

    for (int kt = 0; kt < nkt; ++kt) {
        const int kb0 = kt * 32;

        // K tiles as A operand: lane m = key-in-tile, contiguous over head dim
        const _Float16* kr0 = kbase + (size_t)(kb0 + l16) * DMODEL;
        const _Float16* kr1 = kr0 + (size_t)16 * DMODEL;
        v16h ka00 = combine8(*(const v8h*)(kr0 + kboff),      *(const v8h*)(kr0 + kboff + 16));
        v16h ka01 = combine8(*(const v8h*)(kr0 + 32 + kboff), *(const v8h*)(kr0 + 48 + kboff));
        v16h ka10 = combine8(*(const v8h*)(kr1 + kboff),      *(const v8h*)(kr1 + kboff + 16));
        v16h ka11 = combine8(*(const v8h*)(kr1 + 32 + kboff), *(const v8h*)(kr1 + 48 + kboff));

        v8f s0 = {}, s1 = {};
        s0 = wmma_f16(ka00, qb0, s0);
        s1 = wmma_f16(ka10, qb0, s1);
        s0 = wmma_f16(ka01, qb1, s0);
        s1 = wmma_f16(ka11, qb1, s1);
        // lane holds S^T[key][query]: tile0 key = kb0+8h+r, tile1 key = +16

        // causal mask + per-query max (keys on registers -> VALU reduce)
        float x0[8], x1[8];
        float mx = mrun;
        #pragma unroll
        for (int r = 0; r < 8; ++r) {
            const int k0i = kb0 + kboff + r;
            x0[r] = (k0i      <= qidx) ? s0[r] * scale : -3.0e38f;
            x1[r] = (k0i + 16 <= qidx) ? s1[r] * scale : -3.0e38f;
            mx = fmaxf(mx, fmaxf(x0[r], x1[r]));
        }
        mx = fmaxf(mx, __shfl_xor(mx, 16, 32));   // other half's keys

        const float sc = __expf(mrun - mx);
        mrun = mx;

        // exp -> directly into the P*V A-fragment; accumulate row sum
        v16h pa;
        float sum = 0.0f;
        #pragma unroll
        for (int r = 0; r < 8; ++r) {
            const float p0 = __expf(x0[r] - mx);
            const float p1 = __expf(x1[r] - mx);
            sum += p0 + p1;
            pa[r]     = (_Float16)p0;
            pa[r + 8] = (_Float16)p1;
        }
        sum += __shfl_xor(sum, 16, 32);
        lrun = lrun * sc + sum;

        // rescale O accumulators: query on VGPR index = 8*hlf + r
        #pragma unroll
        for (int r = 0; r < 8; ++r) {
            const float scr = __shfl(sc, kboff + r, 16);
            acc[0][r] *= scr; acc[1][r] *= scr;
            acc[2][r] *= scr; acc[3][r] *= scr;
        }

        // O += P[16x32] x V[32x64]; V^T rows are contiguous over keys
        #pragma unroll
        for (int t = 0; t < 4; ++t) {
            const _Float16* vrow = vtbase + (size_t)(t * 16 + l16) * SEQ + kb0 + kboff;
            v16h vb = combine8(*(const v8h*)(vrow), *(const v8h*)(vrow + 16));
            acc[t] = wmma_f16(pa, vb, acc[t]);
        }
    }

    // epilogue: normalize (broadcast 1/l to O layout) and store f16
    const float linv = 1.0f / lrun;
    #pragma unroll
    for (int r = 0; r < 8; ++r) {
        const float lr  = __shfl(linv, kboff + r, 16);
        const int   row = q0 + kboff + r;
        #pragma unroll
        for (int t = 0; t < 4; ++t)
            o[(size_t)(b * SEQ + row) * DMODEL + h * HDIM + t * 16 + l16] =
                (_Float16)(acc[t][r] * lr);
    }
}

// ---------------------------------------------------------------------------
// Host-side orchestration
// ---------------------------------------------------------------------------
extern "C" void kernel_launch(void* const* d_in, const int* in_sizes, int n_in,
                              void* d_out, int out_size, void* d_ws, size_t ws_size,
                              hipStream_t stream) {
    const float* x        = (const float*)d_in[0];
    // d_in[1] = causal mask (bool) -- causality hardcoded in the kernel
    const float* Wq_down  = (const float*)d_in[2];
    const float* Wkv_down = (const float*)d_in[3];
    const float* Wq_up    = (const float*)d_in[4];
    const float* Wk_up    = (const float*)d_in[5];
    const float* Wv_up    = (const float*)d_in[6];
    const float* Wout     = (const float*)d_in[7];
    float* out = (float*)d_out;

    // carve f16 workspace (256B aligned)
    char* p = (char*)d_ws;
    auto alloc = [&](size_t halves) -> _Float16* {
        _Float16* r = (_Float16*)p;
        p += ((halves * sizeof(_Float16) + 255) / 256) * 256;
        return r;
    };
    _Float16* x16   = alloc((size_t)MTOT * DMODEL);   // reused as attention out
    _Float16* qbuf  = alloc((size_t)MTOT * DMODEL);
    _Float16* kbuf  = alloc((size_t)MTOT * DMODEL);
    _Float16* vtbuf = alloc((size_t)MTOT * DMODEL);   // V transposed [B,H,D,T]
    _Float16* latq  = alloc((size_t)MTOT * DLATENT);
    _Float16* latkv = alloc((size_t)MTOT * DLATENT);
    _Float16* wqd   = alloc((size_t)DLATENT * DMODEL);
    _Float16* wkvd  = alloc((size_t)DLATENT * DMODEL);
    _Float16* wqu   = alloc((size_t)DMODEL * DLATENT);
    _Float16* wku   = alloc((size_t)DMODEL * DLATENT);
    _Float16* wvu   = alloc((size_t)DMODEL * DLATENT);
    _Float16* wo    = alloc((size_t)DMODEL * DMODEL);

    auto cvt = [&](const float* src, _Float16* dst, int n) {
        f32_to_f16_kernel<<<(n + 255) / 256, 256, 0, stream>>>(src, dst, n);
    };
    cvt(x,        x16,  MTOT * DMODEL);
    cvt(Wq_down,  wqd,  DLATENT * DMODEL);
    cvt(Wkv_down, wkvd, DLATENT * DMODEL);
    cvt(Wq_up,    wqu,  DMODEL * DLATENT);
    cvt(Wk_up,    wku,  DMODEL * DLATENT);
    cvt(Wv_up,    wvu,  DMODEL * DLATENT);
    cvt(Wout,     wo,   DMODEL * DMODEL);

    // down-projections: [8192,1024] x [256,1024]^T -> [8192,256]
    // 4 waves x 64 cols = 256 cols per block
    gemm_wmma_kernel<0><<<dim3(MTOT / 16, 1), 128, 0, stream>>>(
        x16, wqd, latq, MTOT, DLATENT, DMODEL);
    gemm_wmma_kernel<0><<<dim3(MTOT / 16, 1), 128, 0, stream>>>(
        x16, wkvd, latkv, MTOT, DLATENT, DMODEL);

    // up-projections: [8192,256] x [1024,256]^T -> [8192,1024]
    // 8 waves x 64 cols = 512 cols per block, grid.y = 2
    gemm_wmma_kernel<0><<<dim3(MTOT / 16, 2), 256, 0, stream>>>(
        latq, wqu, qbuf, MTOT, DMODEL, DLATENT);
    gemm_wmma_kernel<0><<<dim3(MTOT / 16, 2), 256, 0, stream>>>(
        latkv, wku, kbuf, MTOT, DMODEL, DLATENT);
    gemm_wmma_kernel<2><<<dim3(MTOT / 16, 2), 256, 0, stream>>>(
        latkv, wvu, vtbuf, MTOT, DMODEL, DLATENT);   // V stored transposed

    // causal attention; attention output reuses x16 buffer
    _Float16* att = x16;
    mla_attention_kernel<<<dim3(SEQ / 16 / 4, NHEADS, BATCH), 128, 0, stream>>>(
        qbuf, kbuf, vtbuf, att);

    // output projection: [8192,1024] x [1024,1024]^T -> f32 [8192,1024]
    gemm_wmma_kernel<1><<<dim3(MTOT / 16, 2), 256, 0, stream>>>(
        att, wo, out, MTOT, DMODEL, DMODEL);
}
